// QTypeEmbeddingEmbPred_59871844106649
// MI455X (gfx1250) — matrix operand order repaired and checked by reading, not compiled, not run on hardware
//
#include <hip/hip_runtime.h>
#include <hip/hip_bf16.h>
#include <stdint.h>

// Problem constants (match reference)
#define HIDDEN 768
#define VOCAB  2048
#define EMB    3072
#define BATCH  8192

typedef __attribute__((ext_vector_type(16))) __bf16    bf16x16;
typedef __attribute__((ext_vector_type(8)))  float     f32x8;
typedef __attribute__((ext_vector_type(4)))  unsigned  u32x4;
typedef __attribute__((ext_vector_type(4)))  int       i32x4;
typedef __attribute__((ext_vector_type(2)))  float     f32x2;
typedef __attribute__((ext_vector_type(4)))  float     f32x4;

#define AS1 __attribute__((address_space(1)))
#define AS3 __attribute__((address_space(3)))

#if __has_builtin(__builtin_amdgcn_global_load_async_to_lds_b128)
#define HAVE_ASYNC_LDS 1
#else
#define HAVE_ASYNC_LDS 0
#endif

template <int N>
__device__ __forceinline__ void wait_asynccnt() {
#if HAVE_ASYNC_LDS
#if __has_builtin(__builtin_amdgcn_s_wait_asynccnt)
    __builtin_amdgcn_s_wait_asynccnt(N);            // N is an ICE (template param)
#else
    asm volatile("s_wait_asynccnt %0" :: "i"(N) : "memory");
#endif
#endif
}

struct Frag4 { u32x4 q[2]; };   // 32 bytes == bf16x16; loads lower to ds_load_b128

__device__ __forceinline__ unsigned short f2bf(float f) {
    union { float f; unsigned u; } x; x.f = f;
    unsigned r = x.u + 0x7FFFu + ((x.u >> 16) & 1u);  // round-to-nearest-even
    return (unsigned short)(r >> 16);
}
__device__ __forceinline__ unsigned pack2bf(float lo, float hi) {
    return (unsigned)f2bf(lo) | ((unsigned)f2bf(hi) << 16);
}

// ---------------------------------------------------------------------------
// Kernel A: G[v,k] = sum_e W[k,e] * table[v,e]   (bf16 out, v-major [2048,768])
// Block 256 (8 waves). Tile 64(v) x 32(k), e-loop step 32.
// LDS row stride 40 bf16 = 20 dw: uint4-aligned, 20*l mod 64 distinct (no conflicts)
// ---------------------------------------------------------------------------
__global__ __launch_bounds__(256) void build_G(const float* __restrict__ W,
                                               const float* __restrict__ table,
                                               unsigned short* __restrict__ G) {
    const int bx  = blockIdx.x;
    const int v0  = (bx & 31) * 64;   // 32 v-tiles
    const int k0  = (bx >> 5) * 32;   // 24 k-tiles
    const int tid = threadIdx.x;
    const int lane = tid & 31;
    const int wv   = tid >> 5;
    const int vt   = (wv & 3) * 16;
    const int kt   = (wv >> 2) * 16;

    __shared__ __align__(16) unsigned short Tsh[64 * 40];
    __shared__ __align__(16) unsigned short Wsh[32 * 40];
    unsigned* Tsh32 = (unsigned*)Tsh;
    unsigned* Wsh32 = (unsigned*)Wsh;

    f32x8 acc = {};

    const int arow = vt + (lane & 15);
    const int bcol = kt + (lane & 15);
    const int h    = lane >> 4;          // half-wave select

    for (int e0 = 0; e0 < EMB; e0 += 32) {
        #pragma unroll
        for (int i = 0; i < 4; ++i) {                 // table tile 64x32 -> 1024 pairs
            int idx = tid + i * 256;
            int v = idx >> 4, ep = (idx & 15) * 2;
            const f32x2 t = *(const f32x2*)&table[(size_t)(v0 + v) * EMB + e0 + ep];
            Tsh32[v * 20 + (ep >> 1)] = pack2bf(t.x, t.y);
        }
        #pragma unroll
        for (int i = 0; i < 2; ++i) {                 // W tile 32x32 -> 512 pairs
            int idx = tid + i * 256;
            int k = idx >> 4, ep = (idx & 15) * 2;
            const f32x2 w = *(const f32x2*)&W[(size_t)(k0 + k) * EMB + e0 + ep];
            Wsh32[k * 20 + (ep >> 1)] = pack2bf(w.x, w.y);
        }
        __syncthreads();

        const u32x4* T4 = (const u32x4*)Tsh;          // row stride 5 uint4
        const u32x4* W4 = (const u32x4*)Wsh;
        Frag4 fa, fb;
        fa.q[0] = T4[arow * 5 + h];                   // K 0..7  / 8..15
        fa.q[1] = T4[arow * 5 + 2 + h];               // K 16..23 / 24..31
        fb.q[0] = W4[bcol * 5 + h];                   // B[e,n] = W[n][e]
        fb.q[1] = W4[bcol * 5 + 2 + h];

        bf16x16 a = __builtin_bit_cast(bf16x16, fa);
        bf16x16 b = __builtin_bit_cast(bf16x16, fb);
        acc = __builtin_amdgcn_wmma_f32_16x16x32_bf16(false, a, false, b,
                                                      (short)0, acc, false, false);
        __syncthreads();
    }

    const int n = lane & 15;
    #pragma unroll
    for (int j = 0; j < 8; ++j) {
        int v = v0 + vt + j + h * 8;
        int k = k0 + kt + n;
        G[(size_t)v * HIDDEN + k] = f2bf(acc[j]);
    }
}

// ---------------------------------------------------------------------------
// Sb[v] = dot(b_inner, table[v,:])   (one wave per v)
// ---------------------------------------------------------------------------
__global__ __launch_bounds__(256) void build_Sb(const float* __restrict__ bvec,
                                                const float* __restrict__ table,
                                                float* __restrict__ Sb) {
    const int lane = threadIdx.x & 31;
    const int v = blockIdx.x * 8 + (threadIdx.x >> 5);
    float s = 0.f;
    for (int e = lane; e < EMB; e += 32) s += bvec[e] * table[(size_t)v * EMB + e];
    #pragma unroll
    for (int off = 16; off > 0; off >>= 1) s += __shfl_xor(s, off, 32);
    if (lane == 0) Sb[v] = s;
}

// ---------------------------------------------------------------------------
// Kernel B: fused score GEMM + per-row argmax.
// Block 128 (4 waves); 64 rows resident in LDS as bf16; stream 32-column
// chunks of G, DOUBLE-BUFFERED via GLOBAL_LOAD_ASYNC_TO_LDS_B128 (ASYNCcnt),
// two 16-wide N-tiles per chunk -> A-fragment reuse, dual accumulators.
// LDS: Ash 64x776 + 2 x Bsh 32x776 bf16 = 198,656 B (CDNA5 320KB/WGP).
// Row stride 776 bf16 = 388 dw: uint4-aligned, 388 mod 64 = 4 -> conflict-free.
// ---------------------------------------------------------------------------
__global__ __launch_bounds__(128) void score_argmax(const float* __restrict__ raw,
                                                    const unsigned short* __restrict__ G,
                                                    const float* __restrict__ Sb,
                                                    int* __restrict__ idxOut) {
    extern __shared__ char smem[];
    unsigned short* Ash  = (unsigned short*)smem;    // 64 rows, stride 776 bf16
    unsigned short* Bsh0 = Ash + 64 * 776;           // 32 rows, stride 776 bf16
    unsigned short* Bsh1 = Bsh0 + 32 * 776;          // 32 rows, stride 776 bf16
    const int tid = threadIdx.x, lane = tid & 31, wv = tid >> 5;
    const int r0 = blockIdx.x * 64;

    const u32x4* Gq = (const u32x4*)G;               // G row = 96 uint4, 16B aligned
    u32x4* BqBuf[2] = { (u32x4*)Bsh0, (u32x4*)Bsh1 };
    const int srb = tid >> 2;                        // staging: 4 threads per row
    const int sc0 = tid & 3;

    // Issue 24 async b128 copies per thread (per wave: ASYNCcnt += 24).
    auto stage = [&](int buf, int v0c) {
        u32x4* Bq = BqBuf[buf];
        #pragma unroll
        for (int i = 0; i < 24; ++i) {
            int c4 = sc0 + i * 4;
            const u32x4* src = Gq + (size_t)(v0c + srb) * 96 + c4;
            u32x4*       dst = Bq + srb * 97 + c4;
#if HAVE_ASYNC_LDS
            __builtin_amdgcn_global_load_async_to_lds_b128(
                (AS1 i32x4*)(void*)src, (AS3 i32x4*)(void*)dst, 0, 0);
#else
            *dst = *src;
#endif
        }
    };

    stage(0, 0);                                     // prologue: chunk 0 in flight

    // Stage A once: 64 x 768 f32 -> packed bf16 pairs (overlaps with async loads)
    unsigned* Ash32 = (unsigned*)Ash;
    for (int r = 0; r < 64; ++r) {
        #pragma unroll
        for (int i = 0; i < 3; ++i) {
            int c = tid + i * 128;                   // pair index 0..383
            const f32x2 t = *(const f32x2*)&raw[(size_t)(r0 + r) * HIDDEN + 2 * c];
            Ash32[r * 388 + c] = pack2bf(t.x, t.y);
        }
    }

    const u32x4* A4 = (const u32x4*)Ash;             // row stride 97 uint4

    float best[8]; int besti[8];
    #pragma unroll
    for (int j = 0; j < 8; ++j) { best[j] = -3.4e38f; besti[j] = 0; }

    const int arow = wv * 16 + (lane & 15);
    const int bcol = lane & 15;
    const int h    = lane >> 4;

    for (int v0c = 0; v0c < VOCAB; v0c += 32) {
        const int cur = (v0c >> 5) & 1;
        if (v0c + 32 < VOCAB) {
            stage(cur ^ 1, v0c + 32);                // next chunk into other buf
            if (tid < 32 && v0c + 64 < VOCAB)        // L2 hint for chunk after next
                __builtin_prefetch(&Gq[(size_t)(v0c + 64 + tid) * 96], 0, 1);
            wait_asynccnt<24>();                     // chunk v0c complete (in-order)
        } else {
            wait_asynccnt<0>();
        }
        __syncthreads();                             // all waves' chunk data visible

        const float sb0 = Sb[v0c + bcol];
        const float sb1 = Sb[v0c + 16 + bcol];
        const u32x4* B4 = (const u32x4*)BqBuf[cur];
        f32x8 acc0 = {}, acc1 = {};
        #pragma unroll 4
        for (int kk4 = 0; kk4 < 96; kk4 += 4) {      // K=768, step 32 bf16 = 4 uint4
            Frag4 fa, fb0, fb1;
            fa.q[0]  = A4[arow * 97 + kk4 + h];
            fa.q[1]  = A4[arow * 97 + kk4 + 2 + h];
            fb0.q[0] = B4[bcol * 97 + kk4 + h];
            fb0.q[1] = B4[bcol * 97 + kk4 + 2 + h];
            fb1.q[0] = B4[(16 + bcol) * 97 + kk4 + h];
            fb1.q[1] = B4[(16 + bcol) * 97 + kk4 + 2 + h];
            bf16x16 a  = __builtin_bit_cast(bf16x16, fa);
            bf16x16 b0 = __builtin_bit_cast(bf16x16, fb0);
            bf16x16 b1 = __builtin_bit_cast(bf16x16, fb1);
            acc0 = __builtin_amdgcn_wmma_f32_16x16x32_bf16(false, a, false, b0,
                                                           (short)0, acc0, false, false);
            acc1 = __builtin_amdgcn_wmma_f32_16x16x32_bf16(false, a, false, b1,
                                                           (short)0, acc1, false, false);
        }

        // Fused argmax: combine both N-tiles, reduce across 16-lane halves (wave32)
        #pragma unroll
        for (int j = 0; j < 8; ++j) {
            float s = acc0[j] + sb0;
            int ci = v0c + bcol;
            float s1 = acc1[j] + sb1;                // higher index: replace only if >
            if (s1 > s) { s = s1; ci = v0c + 16 + bcol; }
            #pragma unroll
            for (int off = 1; off < 16; off <<= 1) {
                float os = __shfl_xor(s, off, 32);
                int   oi = __shfl_xor(ci, off, 32);
                if (os > s || (os == s && oi < ci)) { s = os; ci = oi; }
            }
            if (s > best[j] || (s == best[j] && ci < besti[j])) { best[j] = s; besti[j] = ci; }
        }
        __syncthreads();                             // reads done before buf reuse
    }

    #pragma unroll
    for (int j = 0; j < 8; ++j) {
        int row = r0 + wv * 16 + j + h * 8;
        if ((lane & 15) == 0) idxOut[row] = besti[j];   // lanes 0 / 16 own the result
    }
}

// ---------------------------------------------------------------------------
// Kernel C: out[r,:] = table[idx[r],:]  (b128 copy, 12KB per row)
// ---------------------------------------------------------------------------
__global__ __launch_bounds__(256) void gather_rows(const float* __restrict__ table,
                                                   const int* __restrict__ idx,
                                                   float* __restrict__ out) {
    const int row = blockIdx.x;
    const int v = idx[row];
    const f32x4* src = (const f32x4*)(table + (size_t)v * EMB);
    f32x4*       dst = (f32x4*)(out + (size_t)row * EMB);
    for (int i = threadIdx.x; i < EMB / 4; i += 256) dst[i] = src[i];
}

// ---------------------------------------------------------------------------
extern "C" void kernel_launch(void* const* d_in, const int* in_sizes, int n_in,
                              void* d_out, int out_size, void* d_ws, size_t ws_size,
                              hipStream_t stream) {
    const float* raw   = (const float*)d_in[0];   // [8192,768]
    const float* W     = (const float*)d_in[1];   // [768,3072]
    const float* bvec  = (const float*)d_in[2];   // [3072]
    const float* table = (const float*)d_in[3];   // [2048,3072]
    float* out = (float*)d_out;                   // [8192,3072]

    char* ws = (char*)d_ws;
    unsigned short* G  = (unsigned short*)ws;                           // 3,145,728 B
    float* Sb = (float*)(ws + (size_t)VOCAB * HIDDEN * 2);              // 8,192 B
    int*  idx = (int*)  (ws + (size_t)VOCAB * HIDDEN * 2 + VOCAB * 4);  // 32,768 B

    build_G<<<32 * 24, 256, 0, stream>>>(W, table, G);
    build_Sb<<<VOCAB / 8, 256, 0, stream>>>(bvec, table, Sb);
    const size_t smem = (size_t)(64 * 776 + 2 * 32 * 776) * 2;  // 198,656 B
    score_argmax<<<BATCH / 64, 128, smem, stream>>>(raw, G, Sb, idx);
    gather_rows<<<BATCH, 256, 0, stream>>>(table, idx, out);
}
